// RGB_SClusterFormer_81535659147850
// MI455X (gfx1250) — compile-verified, matrix operand order checked by reading, not run on hardware
//
#include <hip/hip_runtime.h>
#include <hip/hip_bf16.h>
#include <math.h>

// ---------------------------------------------------------------------------
// Problem constants (from the reference)
// ---------------------------------------------------------------------------
#define Bt_   16
#define PS_   32
#define D_    768
#define HEADS_ 8
#define HD_   24
#define PROP_ 4
#define N_    (PS_*PS_)          // 1024
#define M_    (PROP_*PROP_)      // 16
#define INNER_ (HEADS_*HD_)      // 192
#define HID_  (4*D_)             // 3072
#define R_    (Bt_*N_)           // 16384 rows
#define BE_   (Bt_*HEADS_)       // 128

typedef __attribute__((ext_vector_type(16))) _Float16 v16h;
typedef __attribute__((ext_vector_type(8)))  _Float16 v8h;
typedef __attribute__((ext_vector_type(8)))  float    v8f;

// ---------------------------------------------------------------------------
// f32 -> f16 conversion (weights)
// ---------------------------------------------------------------------------
__global__ __launch_bounds__(256)
void cvt_f16_kernel(const float* __restrict__ in, _Float16* __restrict__ out, int n)
{
    int i = blockIdx.x * 256 + threadIdx.x;
    if (i < n) out[i] = (_Float16)in[i];
}

__global__ __launch_bounds__(256)
void zero_kernel(float* __restrict__ p, int n)
{
    int i = blockIdx.x * 256 + threadIdx.x;
    if (i < n) p[i] = 0.0f;
}

// ---------------------------------------------------------------------------
// LayerNorm row kernel: out_f16[row, :] = (x - mu)/sqrt(var+1e-5) * w + b
// one block (256 threads) per row of D_=768
// ---------------------------------------------------------------------------
__global__ __launch_bounds__(256)
void ln_f16_kernel(const float* __restrict__ x, const float* __restrict__ w,
                   const float* __restrict__ b, _Float16* __restrict__ out)
{
    __shared__ float sred[256];
    const int row = blockIdx.x;
    const float* xr = x + (size_t)row * D_;
    float s = 0.f, s2 = 0.f;
    for (int i = threadIdx.x; i < D_; i += 256) {
        float v = xr[i];
        s += v; s2 += v * v;
    }
    sred[threadIdx.x] = s; __syncthreads();
    for (int off = 128; off; off >>= 1) {
        if (threadIdx.x < off) sred[threadIdx.x] += sred[threadIdx.x + off];
        __syncthreads();
    }
    const float mean = sred[0] * (1.0f / D_);
    __syncthreads();
    sred[threadIdx.x] = s2; __syncthreads();
    for (int off = 128; off; off >>= 1) {
        if (threadIdx.x < off) sred[threadIdx.x] += sred[threadIdx.x + off];
        __syncthreads();
    }
    const float var = sred[0] * (1.0f / D_) - mean * mean;
    const float inv = rsqrtf(var + 1e-5f);
    for (int i = threadIdx.x; i < D_; i += 256) {
        float v = (xr[i] - mean) * inv * w[i] + b[i];
        out[(size_t)row * D_ + i] = (_Float16)v;
    }
}

// ---------------------------------------------------------------------------
// Register-tiled WMMA GEMM:
//   C[Mrows,Ncols](f32) = A[Mrows,K](f16,row-major,lda) x W[Ncols,K]^T + bias
// One wave computes a 32(M) x 64(N) output block = 2x4 grid of 16x16 WMMA
// tiles -> 8 v_wmma per K-step of 32 for 12 b128 loads (A reused across the
// 4 N-tiles, B reused across the 2 M-tiles).
//   A fragment (16x32 f16): lane L (kh=L>>4, m=L&15) holds K = kh*8..+8 and
//     K = kh*8+16..+8 of row m (2 x 16B contiguous loads).
//   B fragment (32x16 f16): lane L holds column n=L&15 over the same K
//     pattern; since B = W^T this is a contiguous chunk of W's row n.
//   C fragment (v8f): acc[r] -> (m = kh*8 + r, n = L&15).
// Epilogues: 0 = f32 store, 1 = f32 store + residual, 2 = GELU -> f16 store
// ---------------------------------------------------------------------------
template<int EPI>
__global__ __launch_bounds__(256)
void wmma_gemm_kernel(const _Float16* __restrict__ A, int lda,
                      const _Float16* __restrict__ W, int ldb,
                      const float* __restrict__ bias,
                      const float* __restrict__ res,
                      float* __restrict__ outF,
                      _Float16* __restrict__ outH,
                      int Ncols, int K)
{
    const int wave = threadIdx.x >> 5;
    const int lane = threadIdx.x & 31;
    const int kh   = lane >> 4;          // which 16-lane half
    const int lrow = lane & 15;
    const int nstrips = Ncols >> 6;      // 64-wide N strips

    const int tile = blockIdx.x * 8 + wave;
    const int tm = tile / nstrips;       // 32-row M block
    const int tn = tile - tm * nstrips;  // 64-col N strip

    const _Float16* a0 = A + (size_t)(tm * 32 + lrow) * lda + kh * 8;
    const _Float16* a1 = a0 + (size_t)16 * lda;
    const _Float16* bp0 = W + (size_t)(tn * 64 +  0 + lrow) * ldb + kh * 8;
    const _Float16* bp1 = W + (size_t)(tn * 64 + 16 + lrow) * ldb + kh * 8;
    const _Float16* bp2 = W + (size_t)(tn * 64 + 32 + lrow) * ldb + kh * 8;
    const _Float16* bp3 = W + (size_t)(tn * 64 + 48 + lrow) * ldb + kh * 8;

    v8f acc[2][4] = {};
#pragma unroll 2
    for (int k = 0; k < K; k += 32) {
        // stream prefetch of upcoming A cachelines (global_prefetch_b8)
        __builtin_prefetch(a0 + k + 256, 0, 0);
        __builtin_prefetch(a1 + k + 256, 0, 0);

        v8h lo, hi;
        v16h av0, av1, bv0, bv1, bv2, bv3;
#define FRAG(dst, ptr)                                                     \
        lo = *(const v8h*)((ptr) + k);                                     \
        hi = *(const v8h*)((ptr) + k + 16);                                \
        _Pragma("unroll")                                                  \
        for (int i_ = 0; i_ < 8; ++i_) { dst[i_] = lo[i_]; dst[i_ + 8] = hi[i_]; }
        FRAG(av0, a0)  FRAG(av1, a1)
        FRAG(bv0, bp0) FRAG(bv1, bp1) FRAG(bv2, bp2) FRAG(bv3, bp3)
#undef FRAG
#define WM(i, j, bv)                                                       \
        acc[i][j] = __builtin_amdgcn_wmma_f32_16x16x32_f16(                \
            false, (i ? av1 : av0), false, bv, (short)0, acc[i][j], false, false);
        WM(0, 0, bv0) WM(0, 1, bv1) WM(0, 2, bv2) WM(0, 3, bv3)
        WM(1, 0, bv0) WM(1, 1, bv1) WM(1, 2, bv2) WM(1, 3, bv3)
#undef WM
    }

#pragma unroll
    for (int j = 0; j < 4; ++j) {
        const int n = tn * 64 + j * 16 + lrow;
        const float bs = bias[n];
#pragma unroll
        for (int i = 0; i < 2; ++i) {
#pragma unroll
            for (int r = 0; r < 8; ++r) {
                const int m = tm * 32 + i * 16 + kh * 8 + r;
                const size_t idx = (size_t)m * Ncols + n;
                float val = acc[i][j][r] + bs;
                if (EPI == 0) {
                    outF[idx] = val;
                } else if (EPI == 1) {
                    outF[idx] = val + res[idx];
                } else { // exact GELU -> f16
                    float g = 0.5f * val * (1.0f + erff(val * 0.70710678118654752f));
                    outH[idx] = (_Float16)g;
                }
            }
        }
    }
}

// ---------------------------------------------------------------------------
// Cluster: pool f,v to PROPxPROP centers, L2-normalize f-centers.
// grid = (BE_, M_), block = 32 (one wave). Lane c < HD_ owns channel c.
// f,v layout: [b*N + n][e*HD + c]  (row-major, stride INNER_)
// ---------------------------------------------------------------------------
__global__ __launch_bounds__(32)
void pool_kernel(const float* __restrict__ f, const float* __restrict__ v,
                 float* __restrict__ cn, float* __restrict__ vcen)
{
    const int be = blockIdx.x, m = blockIdx.y;
    const int b = be >> 3, e = be & 7;
    const int c = threadIdx.x;
    const int py = m >> 2, px = m & 3;   // pooling block in (w,h)
    float fs = 0.f, vs = 0.f;
    if (c < HD_) {
        for (int dy = 0; dy < 8; ++dy)
            for (int dx = 0; dx < 8; ++dx) {
                const int n = (py * 8 + dy) * PS_ + (px * 8 + dx);
                const size_t idx = ((size_t)(b * N_ + n)) * INNER_ + e * HD_ + c;
                fs += f[idx]; vs += v[idx];
            }
        fs *= (1.0f / 64.0f); vs *= (1.0f / 64.0f);
    }
    // wave-wide L2 norm over channels (lanes >= HD_ contribute 0)
    float sq = (c < HD_) ? fs * fs : 0.f;
#pragma unroll
    for (int off = 16; off; off >>= 1) sq += __shfl_xor(sq, off, 32);
    const float nrm = fmaxf(sqrtf(sq), 1e-12f);
    if (c < HD_) {
        const size_t o = ((size_t)be * M_ + m) * HD_ + c;
        cn[o]   = fs / nrm;
        vcen[o] = vs;
    }
}

// ---------------------------------------------------------------------------
// Assignment: per token, normalize f, similarity to 16 centers, leaky-relu'd
// RBF, argmax -> hard assignment; accumulate sim*v into center numerators.
// grid = (BE_, N_/256), block = 256.
// The 16x24-float center table is staged into LDS with gfx1250 async copies
// (global_load_async_to_lds_b128, ASYNCcnt), then consumed after the barrier.
// ---------------------------------------------------------------------------
__global__ __launch_bounds__(256)
void assign_kernel(const float* __restrict__ f, const float* __restrict__ v,
                   const float* __restrict__ cn,
                   const float* __restrict__ alpha, const float* __restrict__ beta,
                   int* __restrict__ amax, float* __restrict__ simv,
                   float* __restrict__ counts, float* __restrict__ outc_num)
{
    __shared__ float scn[M_ * HD_];      // 384 floats = 96 x 16B
    const int be = blockIdx.x;
    const int n  = blockIdx.y * 256 + threadIdx.x;
    const int b = be >> 3, e = be & 7;

    if (threadIdx.x < (M_ * HD_) / 4) {
        unsigned lds_off = (unsigned)(size_t)(&scn[0]) + threadIdx.x * 16u;
        const float* gsrc = cn + (size_t)be * (M_ * HD_) + threadIdx.x * 4;
        asm volatile("global_load_async_to_lds_b128 %0, %1, off"
                     :: "v"(lds_off), "v"(gsrc) : "memory");
    }
    asm volatile("s_wait_asynccnt 0x0" ::: "memory");
    __syncthreads();

    const size_t base = ((size_t)(b * N_ + n)) * INNER_ + e * HD_;
    float xv[HD_];
    float sq = 0.f;
#pragma unroll
    for (int c = 0; c < HD_; ++c) { xv[c] = f[base + c]; sq += xv[c] * xv[c]; }
    const float inv = 1.0f / fmaxf(sqrtf(sq), 1e-12f);
    const float al = alpha[0], bt = beta[0];

    float best = -INFINITY; int bm = 0;
#pragma unroll
    for (int m = 0; m < M_; ++m) {
        float dot = 0.f;
#pragma unroll
        for (int c = 0; c < HD_; ++c) dot += scn[m * HD_ + c] * xv[c];
        dot *= inv;
        float d2 = fmaxf(2.0f - 2.0f * dot, 1e-12f);
        float s = bt + al * __expf(-sqrtf(d2));
        s = (s > 0.f) ? s : 0.2f * s;       // leaky_relu(0.2)
        if (s > best) { best = s; bm = m; } // first max wins (strict >)
    }
    amax[be * N_ + n] = bm;
    simv[be * N_ + n] = best;
    atomicAdd(&counts[be * M_ + bm], 1.0f);
    float* on = &outc_num[((size_t)be * M_ + bm) * HD_];
#pragma unroll
    for (int c = 0; c < HD_; ++c) atomicAdd(&on[c], best * v[base + c]);
}

// outc = (outc_num + vcen) / (counts + 1)
__global__ __launch_bounds__(256)
void finalize_outc_kernel(const float* __restrict__ outc_num,
                          const float* __restrict__ vcen,
                          const float* __restrict__ counts,
                          float* __restrict__ outc)
{
    int i = blockIdx.x * 256 + threadIdx.x;
    if (i < BE_ * M_ * HD_)
        outc[i] = (outc_num[i] + vcen[i]) / (counts[i / HD_] + 1.0f);
}

// Broadcast center values back to tokens, writing the proj-GEMM input (f16)
// in layout [b*N+n][e*HD+c]. grid = (BE_, N_/256), block = 256.
__global__ __launch_bounds__(256)
void broadcast_kernel(const int* __restrict__ amax, const float* __restrict__ simv,
                      const float* __restrict__ outc, _Float16* __restrict__ ch)
{
    const int be = blockIdx.x;
    const int n  = blockIdx.y * 256 + threadIdx.x;
    const int b = be >> 3, e = be & 7;
    const int bm = amax[be * N_ + n];
    const float s = simv[be * N_ + n];
    const float* oc = &outc[((size_t)be * M_ + bm) * HD_];
    _Float16* dst = &ch[((size_t)(b * N_ + n)) * INNER_ + e * HD_];
#pragma unroll
    for (int c = 0; c < HD_; ++c) dst[c] = (_Float16)(s * oc[c]);
}

// ---------------------------------------------------------------------------
// Launch
// ---------------------------------------------------------------------------
extern "C" void kernel_launch(void* const* d_in, const int* in_sizes, int n_in,
                              void* d_out, int out_size, void* d_ws, size_t ws_size,
                              hipStream_t stream)
{
    const float* x      = (const float*)d_in[0];
    const float* ln1_w  = (const float*)d_in[1];
    const float* ln1_b  = (const float*)d_in[2];
    const float* f_w    = (const float*)d_in[3];
    const float* f_b    = (const float*)d_in[4];
    const float* v_w    = (const float*)d_in[5];
    const float* v_b    = (const float*)d_in[6];
    const float* proj_w = (const float*)d_in[7];
    const float* proj_b = (const float*)d_in[8];
    const float* alpha  = (const float*)d_in[9];
    const float* beta   = (const float*)d_in[10];
    const float* ln2_w  = (const float*)d_in[11];
    const float* ln2_b  = (const float*)d_in[12];
    const float* fc1_w  = (const float*)d_in[13];
    const float* fc1_b  = (const float*)d_in[14];
    const float* fc2_w  = (const float*)d_in[15];
    const float* fc2_b  = (const float*)d_in[16];
    float* out = (float*)d_out;

    // ---- workspace carve-out (256B aligned) ----
    char* w = (char*)d_ws;
    auto alloc = [&](size_t bytes) -> char* {
        char* p = w; w += (bytes + 255) & ~(size_t)255; return p;
    };
    _Float16* xh   = (_Float16*)alloc((size_t)R_ * D_ * 2);      // ln1(x) f16; reused for ln2(h)
    _Float16* fwh  = (_Float16*)alloc((size_t)INNER_ * D_ * 2);
    _Float16* vwh  = (_Float16*)alloc((size_t)INNER_ * D_ * 2);
    _Float16* pwh  = (_Float16*)alloc((size_t)D_ * INNER_ * 2);
    _Float16* w1h  = (_Float16*)alloc((size_t)HID_ * D_ * 2);
    _Float16* w2h  = (_Float16*)alloc((size_t)D_ * HID_ * 2);
    float*    fo   = (float*)alloc((size_t)R_ * INNER_ * 4);
    float*    vo   = (float*)alloc((size_t)R_ * INNER_ * 4);
    float*    cn   = (float*)alloc((size_t)BE_ * M_ * HD_ * 4);
    float*    vc   = (float*)alloc((size_t)BE_ * M_ * HD_ * 4);
    float*    on   = (float*)alloc((size_t)BE_ * M_ * HD_ * 4);
    float*    oc   = (float*)alloc((size_t)BE_ * M_ * HD_ * 4);
    float*    ct   = (float*)alloc((size_t)BE_ * M_ * 4);
    int*      am   = (int*)alloc((size_t)BE_ * N_ * 4);
    float*    sv   = (float*)alloc((size_t)BE_ * N_ * 4);
    _Float16* ch   = (_Float16*)alloc((size_t)R_ * INNER_ * 2);
    float*    h    = (float*)alloc((size_t)R_ * D_ * 4);
    _Float16* y1   = (_Float16*)alloc((size_t)R_ * HID_ * 2);
    (void)ws_size; (void)n_in; (void)in_sizes; (void)out_size;

    // ---- 1. weights -> f16 ----
    auto cvt = [&](const float* src, _Float16* dst, int n) {
        cvt_f16_kernel<<<(n + 255) / 256, 256, 0, stream>>>(src, dst, n);
    };
    cvt(f_w,    fwh, INNER_ * D_);
    cvt(v_w,    vwh, INNER_ * D_);
    cvt(proj_w, pwh, D_ * INNER_);
    cvt(fc1_w,  w1h, HID_ * D_);
    cvt(fc2_w,  w2h, D_ * HID_);

    // ---- 2. ln1 ----
    ln_f16_kernel<<<R_, 256, 0, stream>>>(x, ln1_w, ln1_b, xh);

    // ---- 3/4. f and v GEMMs (16384x768 -> 192), f32 out ----
    {
        const int tiles = (R_ / 32) * (INNER_ / 64);   // 32x64 per wave
        wmma_gemm_kernel<0><<<tiles / 8, 256, 0, stream>>>(
            xh, D_, fwh, D_, f_b, nullptr, fo, nullptr, INNER_, D_);
        wmma_gemm_kernel<0><<<tiles / 8, 256, 0, stream>>>(
            xh, D_, vwh, D_, v_b, nullptr, vo, nullptr, INNER_, D_);
    }

    // ---- 5. pool to centers ----
    pool_kernel<<<dim3(BE_, M_), 32, 0, stream>>>(fo, vo, cn, vc);

    // ---- 6. zero accumulators ----
    zero_kernel<<<(BE_ * M_ * HD_ + 255) / 256, 256, 0, stream>>>(on, BE_ * M_ * HD_);
    zero_kernel<<<(BE_ * M_ + 255) / 256, 256, 0, stream>>>(ct, BE_ * M_);

    // ---- 7. assign + 8. finalize + 9. broadcast ----
    assign_kernel<<<dim3(BE_, N_ / 256), 256, 0, stream>>>(fo, vo, cn, alpha, beta,
                                                           am, sv, ct, on);
    finalize_outc_kernel<<<(BE_ * M_ * HD_ + 255) / 256, 256, 0, stream>>>(on, vc, ct, oc);
    broadcast_kernel<<<dim3(BE_, N_ / 256), 256, 0, stream>>>(am, sv, oc, ch);

    // ---- 10. proj GEMM + residual x -> h ----
    {
        const int tiles = (R_ / 32) * (D_ / 64);
        wmma_gemm_kernel<1><<<tiles / 8, 256, 0, stream>>>(
            ch, INNER_, pwh, INNER_, proj_b, x, h, nullptr, D_, INNER_);
    }

    // ---- 11. ln2(h) -> f16 (reuse xh) ----
    ln_f16_kernel<<<R_, 256, 0, stream>>>(h, ln2_w, ln2_b, xh);

    // ---- 12. fc1 GEMM + GELU -> f16 ----
    {
        const int tiles = (R_ / 32) * (HID_ / 64);
        wmma_gemm_kernel<2><<<tiles / 8, 256, 0, stream>>>(
            xh, D_, w1h, D_, fc1_b, nullptr, nullptr, y1, HID_, D_);
    }

    // ---- 13. fc2 GEMM + residual h -> out ----
    {
        const int tiles = (R_ / 32) * (D_ / 64);
        wmma_gemm_kernel<1><<<tiles / 8, 256, 0, stream>>>(
            y1, HID_, w2h, HID_, fc2_b, h, out, nullptr, D_, HID_);
    }
}
